// BahdanauAttention_2808908612183
// MI455X (gfx1250) — matrix operand order repaired and checked by reading, not compiled, not run on hardware
//
#include <hip/hip_runtime.h>
#include <hip/hip_bf16.h>

typedef __attribute__((ext_vector_type(16))) __bf16 v16bf;
typedef __attribute__((ext_vector_type(8)))  float  v8f;

// Problem constants (from reference)
#define NB   32
#define TT   8192
#define CQ_  256
#define CK_  256
#define UU   128

// ---------------------------------------------------------------------------
// Kernel A: qpb[n][u] = b[u] + sum_c (sum_c2 query[n][c2]*w_conv[c][c2]) * w_q[c][u]
// ---------------------------------------------------------------------------
__global__ __launch_bounds__(256) void qpb_kernel(
    const float* __restrict__ query, const float* __restrict__ w_conv,
    const float* __restrict__ w_q, const float* __restrict__ bias,
    float* __restrict__ qpb) {
  __shared__ float qs[CQ_];
  const int n = blockIdx.x;
  const int tid = threadIdx.x;
  const float* qn = query + (size_t)n * CQ_;
  float acc = 0.f;
  const float* wrow = w_conv + (size_t)tid * CQ_;
  for (int c = 0; c < CQ_; ++c) acc += qn[c] * wrow[c];
  qs[tid] = acc;
  __syncthreads();
  if (tid < UU) {
    float a2 = bias[tid];
    for (int c = 0; c < CQ_; ++c) a2 += qs[c] * w_q[(size_t)c * UU + tid];
    qpb[(size_t)n * UU + tid] = a2;
  }
}

// ---------------------------------------------------------------------------
// Helpers for the WMMA score kernel
// ---------------------------------------------------------------------------
__device__ __forceinline__ void load_a_chunks(const float* rowp, int kbase, int h,
                                              float4& f0, float4& f1,
                                              float4& f2, float4& f3) {
  const float4* p1 = (const float4*)(rowp + kbase + h * 16);
  const float4* p2 = (const float4*)(rowp + kbase + h * 16 + 8);
  f0 = p1[0]; f1 = p1[1]; f2 = p2[0]; f3 = p2[1];
}

__device__ __forceinline__ v16bf cvt_a(float4 f0, float4 f1, float4 f2, float4 f3) {
  v16bf a;
  a[0]  = (__bf16)f0.x; a[1]  = (__bf16)f0.y; a[2]  = (__bf16)f0.z; a[3]  = (__bf16)f0.w;
  a[4]  = (__bf16)f1.x; a[5]  = (__bf16)f1.y; a[6]  = (__bf16)f1.z; a[7]  = (__bf16)f1.w;
  a[8]  = (__bf16)f2.x; a[9]  = (__bf16)f2.y; a[10] = (__bf16)f2.z; a[11] = (__bf16)f2.w;
  a[12] = (__bf16)f3.x; a[13] = (__bf16)f3.y; a[14] = (__bf16)f3.z; a[15] = (__bf16)f3.w;
  return a;
}

// Branch-free saturating tanh: 1 - 2/(e^{2x}+1).  Correct limits as x -> +/-inf
// (e->inf gives 1, e->0 gives -1); uses fast v_exp + v_rcp, no exec-mask branches.
__device__ __forceinline__ float fast_tanh(float x) {
  float e = __expf(2.0f * x);
  float r = __builtin_amdgcn_rcpf(e + 1.0f);
  return 1.0f - 2.0f * r;
}

// ---------------------------------------------------------------------------
// Kernel B: scores[n][t] = gumbel[n][t] + sum_u v[u]*tanh(qpb[n][u] + kp[n][t][u])
//           kp = key @ w_k with v_wmma_f32_16x16x32_bf16.
// Block: 256 threads (8 waves). Each wave owns 4 row-tiles of 16 rows.
// grid = (T / 512 = 16, N = 32)
// kk loop kept ROLLED; A globals double-buffered; B LDS loads batched.
// ---------------------------------------------------------------------------
#define TILES_PER_WAVE 4
#define ROWS_PER_BLOCK (8 * TILES_PER_WAVE * 16)  // 512

__global__ __launch_bounds__(256) void score_kernel(
    const float* __restrict__ key, const float* __restrict__ w_k,
    const float* __restrict__ qpb, const float* __restrict__ v,
    const float* __restrict__ gumbel, float* __restrict__ scores) {
  // w_k staged as bf16, fragment-swizzled:
  // bsm[((kc*128 + u) << 4) + e] = w_k[kc*16 + e][u]
  // -> each B fragment (16 contiguous K of one u-column) is one aligned 32B read.
  __shared__ __bf16 bsm[CK_ * UU];  // 64 KB exactly

  const int n    = blockIdx.y;
  const int tid  = threadIdx.x;
  const int lane = tid & 31;
  const int wave = tid >> 5;
  const int c    = lane & 15;   // column / row index within half
  const int h    = lane >> 4;   // half selector

  for (int idx = tid; idx < CK_ * UU; idx += 256) {
    const int u = idx & (UU - 1);
    const int k = idx >> 7;
    const int kc = k >> 4;
    const int e  = k & 15;
    bsm[(((kc << 7) + u) << 4) + e] = (__bf16)w_k[(size_t)k * UU + u];
  }
  __syncthreads();

  float qv[8], vv[8];
#pragma unroll
  for (int j = 0; j < 8; ++j) {
    qv[j] = qpb[(size_t)n * UU + j * 16 + c];
    vv[j] = v[j * 16 + c];
  }

  const float* keyn = key + (size_t)n * TT * CK_;

#pragma unroll 1
  for (int tile = 0; tile < TILES_PER_WAVE; ++tile) {
    const int r0 = blockIdx.x * ROWS_PER_BLOCK + wave * (TILES_PER_WAVE * 16) + tile * 16;
    const float* rowp = keyn + (size_t)(r0 + c) * CK_;

    v8f acc[8] = {};

    // software-pipelined A loads across kk
    float4 f0, f1, f2, f3;
    load_a_chunks(rowp, 0, h, f0, f1, f2, f3);

#pragma unroll 1
    for (int kk = 0; kk < 8; ++kk) {
      // prefetch next A chunk (wraps harmlessly on last iteration)
      float4 g0, g1, g2, g3;
      load_a_chunks(rowp, ((kk + 1) & 7) * 32, h, g0, g1, g2, g3);

      const v16bf a = cvt_a(f0, f1, f2, f3);

      const int kc = kk * 2 + h;
      const __bf16* bbase = &bsm[kc << 11];
      v16bf bf[8];
#pragma unroll
      for (int j = 0; j < 8; ++j)
        bf[j] = *(const v16bf*)&bbase[(j * 16 + c) << 4];
#pragma unroll
      for (int j = 0; j < 8; ++j)
        acc[j] = __builtin_amdgcn_wmma_f32_16x16x32_bf16(
            false, a, false, bf[j], (short)0, acc[j], false, false);

      f0 = g0; f1 = g1; f2 = g2; f3 = g3;
    }

    // Epilogue: score[row] = sum_u v[u]*tanh(kp[row][u] + qpb[u])
    // C layout: lanes 0-15 VGPR r -> (M=r, N=lane); lanes 16-31 -> (M=8+r, N=lane-16)
    float sc[8];
#pragma unroll
    for (int r = 0; r < 8; ++r) {
      float s = 0.f;
#pragma unroll
      for (int j = 0; j < 8; ++j) s += vv[j] * fast_tanh(acc[j][r] + qv[j]);
#pragma unroll
      for (int off = 1; off < 16; off <<= 1) s += __shfl_xor(s, off, 32);
      sc[r] = s;
    }
    if (c == 0) {
#pragma unroll
      for (int r = 0; r < 8; ++r) {
        const int row = r0 + r + 8 * h;
        scores[(size_t)n * TT + row] = sc[r] + gumbel[(size_t)n * TT + row];
      }
    }
  }
}

// ---------------------------------------------------------------------------
// Kernel C: per n — softmax over T, write align; context = align @ key.
// 1024 threads per block, one block per n. Context pass uses float4 streams.
// ---------------------------------------------------------------------------
__global__ __launch_bounds__(1024) void softmax_ctx_kernel(
    const float* __restrict__ scores, const float* __restrict__ key,
    float* __restrict__ out_ctx, float* __restrict__ out_align) {
  __shared__ float als[TT];             // 32 KB
  __shared__ float red[32];
  __shared__ float4 ctxp[16][CK_ / 4];  // 16 KB
  const int n = blockIdx.x;
  const int tid = threadIdx.x;
  const int lane = tid & 31;
  const int wv = tid >> 5;
  const float* sn = scores + (size_t)n * TT;

  // 1) max
  float m = -3.4e38f;
  for (int t = tid; t < TT; t += 1024) m = fmaxf(m, sn[t]);
#pragma unroll
  for (int off = 16; off >= 1; off >>= 1) m = fmaxf(m, __shfl_xor(m, off, 32));
  if (lane == 0) red[wv] = m;
  __syncthreads();
  if (tid < 32) {
    float x = red[tid];
#pragma unroll
    for (int off = 16; off >= 1; off >>= 1) x = fmaxf(x, __shfl_xor(x, off, 32));
    if (tid == 0) red[0] = x;
  }
  __syncthreads();
  m = red[0];
  __syncthreads();

  // 2) exp + sum
  float ssum = 0.f;
  for (int t = tid; t < TT; t += 1024) {
    float e = __expf(sn[t] - m);
    als[t] = e;
    ssum += e;
  }
#pragma unroll
  for (int off = 16; off >= 1; off >>= 1) ssum += __shfl_xor(ssum, off, 32);
  if (lane == 0) red[wv] = ssum;
  __syncthreads();
  if (tid < 32) {
    float x = red[tid];
#pragma unroll
    for (int off = 16; off >= 1; off >>= 1) x += __shfl_xor(x, off, 32);
    if (tid == 0) red[0] = x;
  }
  __syncthreads();
  const float inv = 1.f / red[0];

  // 3) normalize, emit align
  for (int t = tid; t < TT; t += 1024) {
    float a = als[t] * inv;
    als[t] = a;
    out_align[(size_t)n * TT + t] = a;
  }
  __syncthreads();

  // 4) context[n][c] = sum_t align[t]*key[n][t][c], float4-vectorized
  const int cg = tid & 63;      // channel group of 4 (c = cg*4..cg*4+3)
  const int part = tid >> 6;    // 16 t-partitions of 512
  const float4* kn4 = (const float4*)(key + (size_t)n * TT * CK_);
  float4 acc4 = make_float4(0.f, 0.f, 0.f, 0.f);
  const int t0 = part * (TT / 16), t1 = t0 + (TT / 16);
  for (int t = t0; t < t1; ++t) {
    const float w = als[t];
    const float4 kv = kn4[(size_t)t * (CK_ / 4) + cg];
    acc4.x += w * kv.x; acc4.y += w * kv.y;
    acc4.z += w * kv.z; acc4.w += w * kv.w;
  }
  ctxp[part][cg] = acc4;
  __syncthreads();
  if (part == 0) {
    float4 s = ctxp[0][cg];
#pragma unroll
    for (int p = 1; p < 16; ++p) {
      const float4 q = ctxp[p][cg];
      s.x += q.x; s.y += q.y; s.z += q.z; s.w += q.w;
    }
    ((float4*)(out_ctx + (size_t)n * CK_))[cg] = s;
  }
}

// ---------------------------------------------------------------------------
extern "C" void kernel_launch(void* const* d_in, const int* in_sizes, int n_in,
                              void* d_out, int out_size, void* d_ws, size_t ws_size,
                              hipStream_t stream) {
  (void)in_sizes; (void)n_in; (void)out_size; (void)ws_size;
  const float* query  = (const float*)d_in[0];
  const float* key    = (const float*)d_in[1];
  const float* w_conv = (const float*)d_in[2];
  const float* w_q    = (const float*)d_in[3];
  const float* w_k    = (const float*)d_in[4];
  const float* v      = (const float*)d_in[5];
  const float* bias   = (const float*)d_in[6];
  const float* gumbel = (const float*)d_in[7];

  float* out        = (float*)d_out;
  float* out_ctx    = out;                 // N*CK = 8192 floats
  float* out_align  = out + NB * CK_;      // N*T  = 262144 floats

  float* qpb    = (float*)d_ws;            // N*U floats
  float* scores = qpb + NB * UU;           // N*T floats

  qpb_kernel<<<NB, 256, 0, stream>>>(query, w_conv, w_q, bias, qpb);

  dim3 g2(TT / ROWS_PER_BLOCK, NB);        // (16, 32)
  score_kernel<<<g2, 256, 0, stream>>>(key, w_k, qpb, v, gumbel, scores);

  softmax_ctx_kernel<<<NB, 1024, 0, stream>>>(scores, key, out_ctx, out_align);
}
